// SHFGNN_87256555586173
// MI455X (gfx1250) — compile-verified, matrix operand order checked by reading, not run on hardware
//
#include <hip/hip_runtime.h>
#include <hip/hip_bf16.h>
#include <math.h>

// ---------------------------------------------------------------------------
// CDNA5 (gfx1250) WMMA helpers: v_wmma_f32_16x16x32_f16, wave32.
// ---------------------------------------------------------------------------
typedef __attribute__((ext_vector_type(16))) _Float16 v16h;
typedef __attribute__((ext_vector_type(8)))  float    v8f;

constexpr int CB = 2, CS = 320, CIN = 1024, CH = 128, CHEADS = 4, CNC = 16, CMm = 8;

#define WMMA16(a, b, c) __builtin_amdgcn_wmma_f32_16x16x32_f16(false, (a), false, (b), (short)0, (c), false, false)

// K-offset within a 32-wide K chunk for half element e (0..15), per ISA A/B 16-bit layout.
__device__ __forceinline__ int wm_kmap(int lane, int e) {
  int hi = (lane & 16) ? 8 : 0;
  return (e < 8) ? (e + hi) : (e + 8 + hi);
}

template <typename TA>
__device__ __forceinline__ v16h wm_load_a(const TA* A, int lda, int m0, int k0, int Mr, int K) {
  int lane = threadIdx.x & 31;
  int row = m0 + (lane & 15);
  v16h a = {};
  if (row < Mr) {
    const TA* ap = A + (long)row * lda;
#pragma unroll
    for (int e = 0; e < 16; ++e) {
      int k = k0 + wm_kmap(lane, e);
      a[e] = (k < K) ? (_Float16)ap[k] : (_Float16)0.f;
    }
  }
  return a;
}

__device__ __forceinline__ v16h wm_load_b(const float* W, int ldw, int k0, int n0, int K, int N) {
  int lane = threadIdx.x & 31;
  int col = n0 + (lane & 15);
  v16h b = {};
  if (col < N) {
#pragma unroll
    for (int e = 0; e < 16; ++e) {
      int k = k0 + wm_kmap(lane, e);
      b[e] = (k < K) ? (_Float16)W[(long)k * ldw + col] : (_Float16)0.f;
    }
  }
  return b;
}

__device__ __forceinline__ v16h wm_load_a_lds(const _Float16* Sm, int lda, int k0) {
  int lane = threadIdx.x & 31;
  const _Float16* p = Sm + (lane & 15) * lda;
  v16h a;
#pragma unroll
  for (int e = 0; e < 16; ++e) a[e] = p[k0 + wm_kmap(lane, e)];
  return a;
}

__device__ __forceinline__ float silu_f(float x) { return x / (1.f + __expf(-x)); }

// ---------------------------------------------------------------------------
// Generic tiled WMMA GEMM:  C = act(scale*(A@W [+ A2@W2] [+ C_old]) + bias)
// A:[M,K] packed, W:[K,N] with row stride ldw. One wave per 16x16 C tile.
// ---------------------------------------------------------------------------
enum { ACT_NONE = 0, ACT_RELU = 1, ACT_SILU = 2 };

template <typename TA>
__global__ void __launch_bounds__(32) gemm_wmma_k(const TA* A, const TA* A2,
                                                  const float* W, const float* W2,
                                                  const float* bias, float* C,
                                                  int Mr, int K, int K2, int N,
                                                  int ldw, int ldw2, int act, int accum, float scale) {
  int m0 = blockIdx.x * 16, n0 = blockIdx.y * 16;
  v8f c = {};
  for (int k0 = 0; k0 < K; k0 += 32) {
    v16h a = wm_load_a<TA>(A, K, m0, k0, Mr, K);
    v16h b = wm_load_b(W, ldw, k0, n0, K, N);
    c = WMMA16(a, b, c);
  }
  if (A2) {
    for (int k0 = 0; k0 < K2; k0 += 32) {
      v16h a = wm_load_a<TA>(A2, K2, m0, k0, Mr, K2);
      v16h b = wm_load_b(W2, ldw2, k0, n0, K2, N);
      c = WMMA16(a, b, c);
    }
  }
  int lane = threadIdx.x & 31;
  int col = n0 + (lane & 15);
  if (col >= N) return;
  float bv = bias ? bias[col] : 0.f;
  int rbase = (lane & 16) ? 8 : 0;
#pragma unroll
  for (int r = 0; r < 8; ++r) {
    int row = m0 + r + rbase;
    if (row >= Mr) continue;
    float v = c[r];
    if (accum) v += C[(long)row * N + col];
    v = v * scale + bv;
    if (act == ACT_RELU) v = fmaxf(v, 0.f);
    else if (act == ACT_SILU) v = silu_f(v);
    C[(long)row * N + col] = v;
  }
}

// ---------------------------------------------------------------------------
// Conv1d(H,H,5,p=2) + ReLU, c1 stored [B,S,H]
// ---------------------------------------------------------------------------
__global__ void __launch_bounds__(128) conv1_k(const float* h, const float* W, const float* b, float* c1) {
  int bt = blockIdx.y, s = blockIdx.x, oc = threadIdx.x;
  const float* hb = h + (long)bt * CS * CH;
  float acc = b[oc];
  for (int t = 0; t < 5; ++t) {
    int sp = s + t - 2;
    if (sp < 0 || sp >= CS) continue;
    const float* hr = hb + (long)sp * CH;
    const float* wr = W + (long)oc * CH * 5 + t;
    float a = 0.f;
    for (int ic = 0; ic < CH; ++ic) a += hr[ic] * wr[ic * 5];
    acc += a;
  }
  c1[((long)bt * CS + s) * CH + oc] = fmaxf(acc, 0.f);
}

// Conv1d(H,M,3,p=1) + argmax over channels -> motif id
__global__ void __launch_bounds__(32) conv2_argmax_k(const float* c1, const float* W, const float* b, int* motif) {
  int bt = blockIdx.y, s = blockIdx.x;
  __shared__ float ml[CMm];
  int oc = threadIdx.x;
  if (oc < CMm) {
    float acc = b[oc];
    for (int t = 0; t < 3; ++t) {
      int sp = s + t - 1;
      if (sp < 0 || sp >= CS) continue;
      const float* cr = c1 + ((long)bt * CS + sp) * CH;
      const float* wr = W + (long)oc * CH * 3 + t;
      float a = 0.f;
      for (int ic = 0; ic < CH; ++ic) a += cr[ic] * wr[ic * 3];
      acc += a;
    }
    ml[oc] = acc;
  }
  __syncthreads();
  if (oc == 0) {
    int best = 0; float bv = ml[0];
    for (int c = 1; c < CMm; ++c) if (ml[c] > bv) { bv = ml[c]; best = c; }
    motif[(long)bt * CS + s] = best;
  }
}

// ---------------------------------------------------------------------------
// kNN top-9 per query row (one wave per (b,i)); writes knn[b,i,j] in {0,1}.
// ---------------------------------------------------------------------------
__global__ void __launch_bounds__(32) knn_k(const float* x, unsigned char* knn) {
  int b = blockIdx.y, i = blockIdx.x, lane = threadIdx.x;
  const float* xi = x + ((long)b * CS + i) * 3;
  float xi0 = xi[0], xi1 = xi[1], xi2 = xi[2];
  float d[10]; int jj[10];
  for (int t = 0; t < 10; ++t) {
    int j = lane + t * 32;
    if (j < CS) {
      const float* xj = x + ((long)b * CS + j) * 3;
      float dx = xi0 - xj[0], dy = xi1 - xj[1], dz = xi2 - xj[2];
      d[t] = dx * dx + dy * dy + dz * dz; jj[t] = j;
    } else { d[t] = 3.4e38f; jj[t] = 1 << 30; }
  }
  unsigned sel = 0;
  for (int k = 0; k < 9; ++k) {
    float bm = 3.4e38f; int bj = 1 << 30, bt = -1;
    for (int t = 0; t < 10; ++t)
      if (d[t] < bm || (d[t] == bm && jj[t] < bj)) { bm = d[t]; bj = jj[t]; bt = t; }
    float rm = bm; int rj = bj;
    for (int off = 16; off; off >>= 1) {
      float om = __shfl_xor(rm, off, 32);
      int   oj = __shfl_xor(rj, off, 32);
      if (om < rm || (om == rm && oj < rj)) { rm = om; rj = oj; }
    }
    if (bt >= 0 && bj == rj) { sel |= 1u << bt; d[bt] = 3.4e38f; }
  }
  for (int t = 0; t < 10; ++t) {
    int j = lane + t * 32;
    if (j < CS) knn[((long)b * CS + i) * CS + j] = (sel >> t) & 1;
  }
}

// ---------------------------------------------------------------------------
// Per-row attention softmax stats over j (4 waves = 4 heads per block)
// ---------------------------------------------------------------------------
__global__ void __launch_bounds__(128) attprep_k(const float* al, const float* ar,
                                                 const unsigned char* knn, const int* motif,
                                                 float* amax, float* asum) {
  int b = blockIdx.y, i = blockIdx.x;
  int lane = threadIdx.x & 31, h = threadIdx.x >> 5;
  float ai = al[((long)b * CS + i) * 4 + h];
  int mi = motif[(long)b * CS + i];
  const unsigned char* kr = knn + ((long)b * CS + i) * CS;
  const int* mo = motif + (long)b * CS;
  float mx = -1e30f;
  for (int j = lane; j < CS; j += 32) {
    if (!(kr[j] || (mo[j] == mi && j != i))) continue;
    float l = ai + ar[((long)b * CS + j) * 4 + h];
    l = l > 0.f ? l : 0.2f * l;
    mx = fmaxf(mx, l);
  }
  for (int off = 16; off; off >>= 1) mx = fmaxf(mx, __shfl_xor(mx, off, 32));
  float sum = 0.f;
  for (int j = lane; j < CS; j += 32) {
    if (!(kr[j] || (mo[j] == mi && j != i))) continue;
    float l = ai + ar[((long)b * CS + j) * 4 + h];
    l = l > 0.f ? l : 0.2f * l;
    sum += __expf(l - mx);
  }
  for (int off = 16; off; off >>= 1) sum += __shfl_xor(sum, off, 32);
  if (lane == 0) {
    amax[((long)b * CS + i) * 4 + h] = mx;
    asum[((long)b * CS + i) * 4 + h] = sum;
  }
}

// ---------------------------------------------------------------------------
// Fused shallow message-passing layer. One block per (b, 16-wide j tile).
// For each i: build silu(pre) tile, WMMA msgs GEMM, alpha-weighted h_aggr
// accumulation, coordinate-gate WMMA + x update accumulation.
// ---------------------------------------------------------------------------
__global__ void __launch_bounds__(128) mp_layer_k(
    const float* __restrict__ hAbuf, const float* __restrict__ hBbuf,
    const float* __restrict__ al, const float* __restrict__ ar,
    const float* __restrict__ amax, const float* __restrict__ asum,
    const float* __restrict__ x_in, const unsigned char* __restrict__ knn,
    const int* __restrict__ motif, const float* __restrict__ edge_emb,
    const float* __restrict__ eW1, const float* __restrict__ eb1,
    const float* __restrict__ eW2, const float* __restrict__ eb2,
    const float* __restrict__ xW1, const float* __restrict__ xb1,
    const float* __restrict__ xW2,
    float* __restrict__ h_aggr, float* __restrict__ x_out) {
  int b = blockIdx.y, j0 = blockIdx.x * 16;
  int tid = threadIdx.x, lane = tid & 31, wave = tid >> 5;
  int rbase = (lane & 16) ? 8 : 0;

  __shared__ float    sHB[16 * 128];
  __shared__ _Float16 sA[16 * 128];
  __shared__ float    sMsg[16 * 128];
  __shared__ _Float16 sMM[16 * 32];
  __shared__ float    sWsq[128], sB1[128], sW1e[4 * 128];
  __shared__ float    sSq[16], sMaskJ[16], sEA[16 * 4], sXj[16 * 3], sXacc[16 * 3], sDen[16], sArj[16 * 4], sCM[16];
  __shared__ int      sMj[16];

  for (int idx = tid; idx < 16 * 128; idx += 128) {
    int j = idx >> 7, k = idx & 127;
    sHB[idx] = hBbuf[((long)b * CS + j0 + j) * CH + k];
  }
  sWsq[tid] = eW1[256 * 128 + tid];
  sB1[tid]  = eb1[tid];
  for (int idx = tid; idx < 512; idx += 128) sW1e[idx] = eW1[257 * 128 + idx];
  if (tid < 16) {
    int j = j0 + tid;
    sMj[tid] = motif[(long)b * CS + j];
    for (int c = 0; c < 3; ++c) { sXj[tid * 3 + c] = x_in[((long)b * CS + j) * 3 + c]; sXacc[tid * 3 + c] = 0.f; }
    for (int h = 0; h < 4; ++h) sArj[tid * 4 + h] = ar[((long)b * CS + j) * 4 + h];
    sDen[tid] = 0.f;
  }
  __syncthreads();

  // Preload B-fragments in registers: each wave owns N-tiles wave*2, wave*2+1.
  v16h bW2[2][4], bX1[2];
  float eb2v[2], xb1v[2], xw2v[2];
  int colv[2];
#pragma unroll
  for (int nt = 0; nt < 2; ++nt) {
    int n0 = (wave * 2 + nt) * 16;
#pragma unroll
    for (int kt = 0; kt < 4; ++kt) bW2[nt][kt] = wm_load_b(eW2, 128, kt * 32, n0, 128, 128);
    bX1[nt] = wm_load_b(xW1, 128, 0, n0, 32, 128);
    int col = n0 + (lane & 15);
    colv[nt] = col; eb2v[nt] = eb2[col]; xb1v[nt] = xb1[col]; xw2v[nt] = xW2[col];
  }
  v8f hacc[2] = {};

  for (int i = 0; i < CS; ++i) {
    __syncthreads();
    long ibase = (long)b * CS + i;
    // 1: per-j pair scalars
    if (tid < 16) {
      int j = j0 + tid;
      float xi0 = x_in[ibase * 3], xi1 = x_in[ibase * 3 + 1], xi2 = x_in[ibase * 3 + 2];
      float dx = xi0 - sXj[tid * 3], dy = xi1 - sXj[tid * 3 + 1], dz = xi2 - sXj[tid * 3 + 2];
      sSq[tid] = dx * dx + dy * dy + dz * dz;
      bool kn = knn[ibase * CS + j] != 0;
      bool mm = (motif[ibase] == sMj[tid]) && (j != i);
      sMaskJ[tid] = (kn || mm) ? 1.f : 0.f;
      int et = (kn && mm) ? 2 : (mm ? 1 : 0);
      for (int c = 0; c < 4; ++c) sEA[tid * 4 + c] = edge_emb[et * 4 + c];
      sCM[tid] = 0.f;
    }
    __syncthreads();
    // 2: silu(pre) tile -> f16 LDS (thread t: row tid/8, 16 k's)
    {
      int j = tid >> 3, kb = (tid & 7) * 16;
      const float* hAi = hAbuf + ibase * CH;
      float sq = sSq[j];
      float e0 = sEA[j * 4], e1 = sEA[j * 4 + 1], e2 = sEA[j * 4 + 2], e3 = sEA[j * 4 + 3];
#pragma unroll
      for (int kk = 0; kk < 16; ++kk) {
        int k = kb + kk;
        float p = hAi[k] + sHB[j * 128 + k] + sq * sWsq[k]
                + e0 * sW1e[k] + e1 * sW1e[128 + k] + e2 * sW1e[256 + k] + e3 * sW1e[384 + k]
                + sB1[k];
        sA[j * 128 + k] = (_Float16)silu_f(p);
      }
    }
    __syncthreads();
    // 3: msgs = silu(pre) @ e_W2 + e_b2 ; fused alpha-weighted h_aggr
    {
      v16h aK[4];
#pragma unroll
      for (int kt = 0; kt < 4; ++kt) aK[kt] = wm_load_a_lds(sA, 128, kt * 32);
#pragma unroll
      for (int nt = 0; nt < 2; ++nt) {
        v8f c = {};
#pragma unroll
        for (int kt = 0; kt < 4; ++kt) c = WMMA16(aK[kt], bW2[nt][kt], c);
        int col = colv[nt], hd = col >> 5;
        float ai  = al[ibase * 4 + hd];
        float mxv = amax[ibase * 4 + hd];
        float smv = asum[ibase * 4 + hd];
        float inv = smv > 0.f ? 1.f / smv : 0.f;
#pragma unroll
        for (int r = 0; r < 8; ++r) {
          int jr = r + rbase;
          float m = c[r] + eb2v[nt];
          sMsg[jr * 128 + col] = m;
          float l = ai + sArj[jr * 4 + hd];
          l = l > 0.f ? l : 0.2f * l;
          float alpha = sMaskJ[jr] * __expf(l - mxv) * inv;
          hacc[nt][r] += alpha * m;
        }
      }
    }
    __syncthreads();
    // 4: mean over heads -> [16,32] f16
    for (int idx = tid; idx < 512; idx += 128) {
      int j = idx >> 5, hd = idx & 31;
      float m = 0.25f * (sMsg[j * 128 + hd] + sMsg[j * 128 + 32 + hd] +
                         sMsg[j * 128 + 64 + hd] + sMsg[j * 128 + 96 + hd]);
      sMM[j * 32 + hd] = (_Float16)m;
    }
    __syncthreads();
    // 5: cm = silu(mm @ x_W1 + x_b1) . x_W2
    {
      v16h amm = wm_load_a_lds(sMM, 32, 0);
#pragma unroll
      for (int nt = 0; nt < 2; ++nt) {
        v8f c = {};
        c = WMMA16(amm, bX1[nt], c);
        float part[8];
#pragma unroll
        for (int r = 0; r < 8; ++r) part[r] = silu_f(c[r] + xb1v[nt]) * xw2v[nt];
#pragma unroll
        for (int off = 8; off; off >>= 1) {
#pragma unroll
          for (int r = 0; r < 8; ++r) part[r] += __shfl_xor(part[r], off, 16);
        }
        if ((lane & 15) == 0) {
#pragma unroll
          for (int r = 0; r < 8; ++r) atomicAdd(&sCM[rbase + r], part[r]);
        }
      }
    }
    __syncthreads();
    // 6: x-update accumulation over i
    if (tid < 16) {
      float msk = sMaskJ[tid], cmv = sCM[tid];
      float xi0 = x_in[ibase * 3], xi1 = x_in[ibase * 3 + 1], xi2 = x_in[ibase * 3 + 2];
      sXacc[tid * 3]     += msk * (xi0 - sXj[tid * 3])     * cmv;
      sXacc[tid * 3 + 1] += msk * (xi1 - sXj[tid * 3 + 1]) * cmv;
      sXacc[tid * 3 + 2] += msk * (xi2 - sXj[tid * 3 + 2]) * cmv;
      sDen[tid] += msk;
    }
  }
  __syncthreads();
#pragma unroll
  for (int nt = 0; nt < 2; ++nt) {
    int col = colv[nt];
#pragma unroll
    for (int r = 0; r < 8; ++r) {
      int j = j0 + r + rbase;
      h_aggr[((long)b * CS + j) * CH + col] = hacc[nt][r];
    }
  }
  if (tid < 16) {
    int j = j0 + tid;
    float den = fmaxf(sDen[tid], 1.f);
    for (int c = 0; c < 3; ++c)
      x_out[((long)b * CS + j) * 3 + c] = sXj[tid * 3 + c] + sXacc[tid * 3 + c] / den;
  }
}

// ---------------------------------------------------------------------------
// GATv2: per (b,i) edge scores + softmax over j -> ga (f16, [B,4,S,S])
// ---------------------------------------------------------------------------
__global__ void __launch_bounds__(128) gat_att_k(const float* sl, const float* sr, const float* att,
                                                 const unsigned char* knn, const int* motif,
                                                 _Float16* ga) {
  int b = blockIdx.y, i = blockIdx.x;
  int t = threadIdx.x, lane = t & 31, w = t >> 5;
  __shared__ float sSr[512], sAtt[512], sE[CS * 4];
  __shared__ unsigned char sMk[CS];
  for (int idx = t; idx < 512; idx += 128) {
    sSr[idx] = sr[((long)b * CS + i) * 512 + idx];
    sAtt[idx] = att[idx];
  }
  __syncthreads();
  int mi = motif[(long)b * CS + i];
  for (int j = t; j < CS; j += 128) {
    bool mask = (j == i) || (knn[((long)b * CS + j) * CS + i] != 0) ||
                (motif[(long)b * CS + j] == mi && j != i);
    sMk[j] = mask ? 1 : 0;
    if (mask) {
      const float* slj = sl + ((long)b * CS + j) * 512;
      for (int h = 0; h < 4; ++h) {
        float acc = 0.f;
        for (int k = 0; k < 128; ++k) {
          float v = sSr[h * 128 + k] + slj[h * 128 + k];
          v = v > 0.f ? v : 0.2f * v;
          acc += v * sAtt[h * 128 + k];
        }
        sE[j * 4 + h] = acc;
      }
    }
  }
  __syncthreads();
  int h = w;
  float mx = -1e30f;
  for (int j = lane; j < CS; j += 32) if (sMk[j]) mx = fmaxf(mx, sE[j * 4 + h]);
  for (int off = 16; off; off >>= 1) mx = fmaxf(mx, __shfl_xor(mx, off, 32));
  float sum = 0.f;
  for (int j = lane; j < CS; j += 32) if (sMk[j]) sum += __expf(sE[j * 4 + h] - mx);
  for (int off = 16; off; off >>= 1) sum += __shfl_xor(sum, off, 32);
  float inv = sum > 0.f ? 1.f / sum : 0.f;
  _Float16* gr = ga + (((long)b * 4 + h) * CS + i) * CS;
  for (int j = lane; j < CS; j += 32)
    gr[j] = (_Float16)(sMk[j] ? __expf(sE[j * 4 + h] - mx) * inv : 0.f);
}

// Row softmax over NC=16 assignments; writes transposed s_soft^T [B,NC,S]
__global__ void smax_s_k(const float* s, float* sT) {
  int r = blockIdx.x * blockDim.x + threadIdx.x;
  if (r >= CB * CS) return;
  int b = r / CS, sr_ = r % CS;
  const float* row = s + (long)r * CNC;
  float mx = row[0];
  for (int c = 1; c < CNC; ++c) mx = fmaxf(mx, row[c]);
  float e[CNC], sum = 0.f;
  for (int c = 0; c < CNC; ++c) { e[c] = __expf(row[c] - mx); sum += e[c]; }
  float inv = 1.f / sum;
  for (int c = 0; c < CNC; ++c) sT[((long)b * CNC + c) * CS + sr_] = e[c] * inv;
}

// LayerNorm over H=128; optional second output copy
__global__ void __launch_bounds__(128) ln_k(const float* in, const float* g, const float* be,
                                            float* out, float* out2) {
  long row = blockIdx.x;
  int t = threadIdx.x, lane = t & 31, w = t >> 5;
  __shared__ float red[4];
  float v = in[row * CH + t];
  float s = v;
  for (int off = 16; off; off >>= 1) s += __shfl_xor(s, off, 32);
  if (lane == 0) red[w] = s;
  __syncthreads();
  float mu = (red[0] + red[1] + red[2] + red[3]) * (1.f / 128.f);
  __syncthreads();
  float d = v - mu, s2 = d * d;
  for (int off = 16; off; off >>= 1) s2 += __shfl_xor(s2, off, 32);
  if (lane == 0) red[w] = s2;
  __syncthreads();
  float var = (red[0] + red[1] + red[2] + red[3]) * (1.f / 128.f);
  float o = d * rsqrtf(var + 1e-5f) * g[t] + be[t];
  out[row * CH + t] = o;
  if (out2) out2[row * CH + t] = o;
}

// ---------------------------------------------------------------------------
// Host orchestration
// ---------------------------------------------------------------------------
extern "C" void kernel_launch(void* const* d_in, const int* in_sizes, int n_in,
                              void* d_out, int out_size, void* d_ws, size_t ws_size,
                              hipStream_t stream) {
  (void)in_sizes; (void)n_in; (void)out_size; (void)ws_size;
  auto F = [&](int i) { return (const float*)d_in[i]; };
  // jax pytree leaf order (dict keys sorted):
  const float* h_in   = F(0);
  const float* c1W    = F(1);  const float* c1b = F(2);
  const float* c2W    = F(3);  const float* c2b = F(4);
  const float* coordW = F(5);  const float* coordb = F(6);
  const float* dW1[2] = {F(7), F(11)};
  const float* dW2[2] = {F(8), F(12)};
  const float* db1[2] = {F(9), F(13)};
  const float* db2[2] = {F(10), F(14)};
  const float* edge_emb = F(15);
  const float* featW  = F(16); const float* featb = F(17);
  const float* gWl = F(18), *gWr = F(19), *gatt = F(20), *gbias = F(21), *gbl = F(22), *gbr = F(23);
  const float* ln_d_b = F(24), *ln_d_g = F(25), *ln_s_b = F(26), *ln_s_g = F(27);
  const float* poolW = F(28), *poolb = F(29);
  struct SL { const float *attW, *eW1, *eW2, *eb1, *eb2, *hW1, *hW2, *hb1, *hb2, *xW1, *xW2, *xb1; };
  SL sls[3];
  for (int L = 0; L < 3; ++L) {
    int b0 = 30 + 12 * L;
    sls[L] = {F(b0), F(b0 + 1), F(b0 + 2), F(b0 + 3), F(b0 + 4), F(b0 + 5),
              F(b0 + 6), F(b0 + 7), F(b0 + 8), F(b0 + 9), F(b0 + 10), F(b0 + 11)};
  }

  // workspace bump allocator
  char* base = (char*)d_ws;
  size_t off = 0;
  auto alloc = [&](size_t bytes) { void* p = base + off; off = (off + bytes + 255) & ~(size_t)255; return p; };
  const int BS = CB * CS;
  float* h     = (float*)alloc((size_t)BS * CH * 4);
  float* x0    = (float*)alloc((size_t)BS * 3 * 4);
  float* x1    = (float*)alloc((size_t)BS * 3 * 4);
  float* c1buf = (float*)alloc((size_t)BS * CH * 4);
  int*   motif = (int*)alloc((size_t)BS * 4);
  unsigned char* knn = (unsigned char*)alloc((size_t)CB * CS * CS);
  float* albuf = (float*)alloc((size_t)BS * 4 * 4);
  float* arbuf = (float*)alloc((size_t)BS * 4 * 4);
  float* amax  = (float*)alloc((size_t)BS * 4 * 4);
  float* asum  = (float*)alloc((size_t)BS * 4 * 4);
  float* hA    = (float*)alloc((size_t)BS * CH * 4);
  float* hB    = (float*)alloc((size_t)BS * CH * 4);
  float* haggr = (float*)alloc((size_t)BS * CH * 4);
  float* hu    = (float*)alloc((size_t)BS * CH * 4);
  float* hsn   = (float*)alloc((size_t)BS * CH * 4);
  float* slb   = (float*)alloc((size_t)BS * 512 * 4);
  float* srb   = (float*)alloc((size_t)BS * 512 * 4);
  _Float16* ga = (_Float16*)alloc((size_t)CB * 4 * CS * CS * 2);
  float* z     = (float*)alloc((size_t)BS * CH * 4);
  float* sbuf  = (float*)alloc((size_t)BS * CNC * 4);
  float* sT    = (float*)alloc((size_t)CB * CNC * CS * 4);
  float* hpool = (float*)alloc((size_t)CB * CNC * CH * 4);
  float* dtmp  = (float*)alloc((size_t)CB * CNC * CH * 4);
  float* hdeep = (float*)alloc((size_t)BS * CH * 4);

  auto gemm = [&](const float* A, const float* A2, const float* W, const float* W2,
                  const float* bias, float* C, int M, int K, int K2, int N,
                  int ldw, int ldw2, int act, int accum, float scale) {
    dim3 grid((M + 15) / 16, (N + 15) / 16);
    gemm_wmma_k<float><<<grid, dim3(32), 0, stream>>>(A, A2, W, W2, bias, C, M, K, K2, N,
                                                      ldw, ldw2, act, accum, scale);
  };
  auto gemm16 = [&](const _Float16* A, const float* W, const float* bias, float* C,
                    int M, int K, int N, int ldw, int accum, float scale) {
    dim3 grid((M + 15) / 16, (N + 15) / 16);
    gemm_wmma_k<_Float16><<<grid, dim3(32), 0, stream>>>(A, (const _Float16*)nullptr, W, nullptr,
                                                         bias, C, M, K, 0, N, ldw, 0, ACT_NONE, accum, scale);
  };

  // projections
  gemm(h_in, nullptr, featW, nullptr, featb, h, BS, CIN, 0, CH, CH, 0, ACT_NONE, 0, 1.f);
  gemm(h_in, nullptr, coordW, nullptr, coordb, x0, BS, CIN, 0, 3, 3, 0, ACT_NONE, 0, 1.f);
  // motif detector + graph construction
  conv1_k<<<dim3(CS, CB), 128, 0, stream>>>(h, c1W, c1b, c1buf);
  conv2_argmax_k<<<dim3(CS, CB), 32, 0, stream>>>(c1buf, c2W, c2b, motif);
  knn_k<<<dim3(CS, CB), 32, 0, stream>>>(x0, knn);

  // shallow layers
  float* xp[4] = {x0, x1, x0, x1};
  for (int L = 0; L < 3; ++L) {
    const SL& sp = sls[L];
    gemm(h, nullptr, sp.attW, nullptr, nullptr, albuf, BS, CH, 0, 4, 4, 0, ACT_NONE, 0, 1.f);
    gemm(h, nullptr, sp.attW + CH * 4, nullptr, nullptr, arbuf, BS, CH, 0, 4, 4, 0, ACT_NONE, 0, 1.f);
    gemm(h, nullptr, sp.eW1, nullptr, nullptr, hA, BS, CH, 0, CH, CH, 0, ACT_NONE, 0, 1.f);
    gemm(h, nullptr, sp.eW1 + CH * CH, nullptr, nullptr, hB, BS, CH, 0, CH, CH, 0, ACT_NONE, 0, 1.f);
    attprep_k<<<dim3(CS, CB), 128, 0, stream>>>(albuf, arbuf, knn, motif, amax, asum);
    mp_layer_k<<<dim3(CS / 16, CB), 128, 0, stream>>>(hA, hB, albuf, arbuf, amax, asum,
        xp[L], knn, motif, edge_emb, sp.eW1, sp.eb1, sp.eW2, sp.eb2,
        sp.xW1, sp.xb1, sp.xW2, haggr, xp[L + 1]);
    gemm(h, haggr, sp.hW1, sp.hW1 + CH * CH, sp.hb1, hu, BS, CH, CH, CH, CH, CH, ACT_SILU, 0, 1.f);
    gemm(hu, nullptr, sp.hW2, nullptr, sp.hb2, h, BS, CH, 0, CH, CH, 0, ACT_NONE, 1, 1.f);
  }

  // h_sn = LN(h)  (also output 0)
  ln_k<<<BS, 128, 0, stream>>>(h, ln_s_g, ln_s_b, hsn, (float*)d_out);

  // GATv2 pooling embed
  gemm(hsn, nullptr, gWl, nullptr, gbl, slb, BS, CH, 0, 512, 512, 0, ACT_NONE, 0, 1.f);
  gemm(hsn, nullptr, gWr, nullptr, gbr, srb, BS, CH, 0, 512, 512, 0, ACT_NONE, 0, 1.f);
  gat_att_k<<<dim3(CS, CB), 128, 0, stream>>>(slb, srb, gatt, knn, motif, ga);
  for (int b = 0; b < CB; ++b)
    for (int hh = 0; hh < 4; ++hh) {
      const _Float16* Ag = ga + (((long)b * 4 + hh) * CS * CS);
      const float* Wg = slb + (long)b * CS * 512 + hh * 128;
      gemm16(Ag, Wg, (hh == 3) ? gbias : nullptr, z + (long)b * CS * CH,
             CS, CS, CH, 512, hh > 0, (hh == 3) ? 0.25f : 1.f);
    }

  // assignment + diff-pool
  gemm(z, nullptr, poolW, nullptr, poolb, sbuf, BS, CH, 0, CNC, CNC, 0, ACT_NONE, 0, 1.f);
  smax_s_k<<<(BS + 63) / 64, 64, 0, stream>>>(sbuf, sT);
  for (int b = 0; b < CB; ++b)
    gemm(sT + (long)b * CNC * CS, nullptr, hsn + (long)b * CS * CH, nullptr, nullptr,
         hpool + (long)b * CNC * CH, CNC, CS, 0, CH, CH, 0, ACT_NONE, 0, 1.f);
  for (int l = 0; l < 2; ++l) {
    gemm(hpool, hpool, dW1[l], dW1[l] + CH * CH, db1[l], dtmp, CB * CNC, CH, CH, CH, CH, CH, ACT_SILU, 0, 1.f);
    gemm(dtmp, nullptr, dW2[l], nullptr, db2[l], hpool, CB * CNC, CH, 0, CH, CH, 0, ACT_NONE, 0, 1.f);
  }
  for (int b = 0; b < CB; ++b)
    gemm(sbuf + (long)b * CS * CNC, nullptr, hpool + (long)b * CNC * CH, nullptr, nullptr,
         hdeep + (long)b * CS * CH, CS, CNC, 0, CH, CH, 0, ACT_NONE, 0, 1.f);

  // output 1: LN(h_deep)
  ln_k<<<BS, 128, 0, stream>>>(hdeep, ln_d_g, ln_d_b, (float*)d_out + (long)BS * CH, nullptr);
}
